// FourierCrossAttention_47717086658904
// MI455X (gfx1250) — matrix operand order repaired and checked by reading, not compiled, not run on hardware
//
#include <hip/hip_runtime.h>
#include <hip/hip_bf16.h>
#include <math.h>

// Problem constants (match reference)
#define BB 16
#define LL 2048
#define HH 8
#define EE 64
#define MM 64

typedef __bf16 bf16_t;
typedef __attribute__((ext_vector_type(16))) __bf16 v16bf;
typedef __attribute__((ext_vector_type(8)))  __bf16 v8bf;
typedef __attribute__((ext_vector_type(8)))  float  v8f;

static __device__ __forceinline__ bf16_t f2b(float f) { return (bf16_t)f; }

static __device__ __forceinline__ v8f wmma_bf16(v16bf a, v16bf b, v8f c) {
  // (neg_a, A, neg_b, B, c_mod, C, reuse_a, reuse_b)
  return __builtin_amdgcn_wmma_f32_16x16x32_bf16(false, a, false, b, (short)0, c,
                                                 false, false);
}

union V16U { v16bf v; v8bf h[2]; };

// A fragment (16x32 bf16): row-major source [row][k], stride in elements.
// lane<16: row=lane, K 0-15 ; lane>=16: row=lane-16, K 16-31.
static __device__ __forceinline__ v16bf load_a_frag(const bf16_t* base, int stride, int lane) {
  const int row = lane & 15;
  const int kb  = (lane < 16) ? 0 : 16;
  const bf16_t* p = base + row * stride + kb;
  V16U u;
  u.h[0] = *(const v8bf*)(p);
  u.h[1] = *(const v8bf*)(p + 8);
  return u.v;
}

// B fragment (32x16 bf16): source stored column-major [n][k], stride in elements.
// lane<16: col=lane, K 0-15 ; lane>=16: col=lane-16, K 16-31.
static __device__ __forceinline__ v16bf load_b_frag(const bf16_t* base, int stride, int lane) {
  const int col = lane & 15;
  const int kb  = (lane < 16) ? 0 : 16;
  return *(const v16bf*)(base + col * stride + kb);
}

// ---------------------------------------------------------------- tables ----
// Forward DFT twiddles, stored [m][t] (B-fragment friendly): 64 x 2048 bf16.
__global__ void __launch_bounds__(256) k_fwd_tables(bf16_t* WcT, bf16_t* WsT) {
  const int idx = blockIdx.x * 256 + threadIdx.x;
  if (idx >= MM * LL) return;
  const int m = idx / LL, t = idx % LL;
  const float th = (float)(2.0 * 3.14159265358979323846 *
                           (double)((m * t) & (LL - 1)) / (double)LL);
  WcT[idx] = f2b(cosf(th));
  WsT[idx] = f2b(-sinf(th));   // Fim = sum x[t] * (-sin)
}

// Inverse DFT tables, stored [t][x]: 2048 x 64 bf16.
// y[t] = sum_x Re[x]*w_x*cos(th) + Im[x]*(-w_x*sin(th)), w_x = (x?2:1)/L
__global__ void __launch_bounds__(256) k_inv_tables(bf16_t* ICt, bf16_t* ISt) {
  const int idx = blockIdx.x * 256 + threadIdx.x;
  if (idx >= LL * MM) return;
  const int t = idx / MM, x = idx % MM;
  const float w = (x == 0 ? 1.0f : 2.0f) / (float)LL;
  const float th = (float)(2.0 * 3.14159265358979323846 *
                           (double)((x * t) & (LL - 1)) / (double)LL);
  ICt[idx] = f2b(w * cosf(th));
  ISt[idx] = f2b(-w * sinf(th));
}

// Weights [H][E][O][M] f32 -> bf16 [h][x][o][e] (B-fragment friendly per (h,x)).
__global__ void __launch_bounds__(256) k_w_transpose(const float* Wr, const float* Wi,
                                                     bf16_t* WtR, bf16_t* WtI) {
  const int idx = blockIdx.x * 256 + threadIdx.x;
  if (idx >= HH * EE * EE * MM) return;
  const int x = idx % MM;
  const int o = (idx / MM) % EE;
  const int e = (idx / (MM * EE)) % EE;
  const int h = idx / (MM * EE * EE);
  const int dst = ((h * MM + x) * EE + o) * EE + e;
  WtR[dst] = f2b(Wr[idx]);
  WtI[dst] = f2b(Wi[idx]);
}

// ------------------------------------------------------ forward DFT GEMM ----
// For each (b,h,tensor): F[e,m] = sum_t X[b,t,h,e] * (cos - i sin).
// Output layouts: tensor 0/1 (q,k) -> [m][e]; tensor 2 (v) -> [e][m].
__global__ void __launch_bounds__(128) k_fdft(const float* q, const float* k, const float* v,
                                              const bf16_t* WcT, const bf16_t* WsT,
                                              bf16_t* QfR, bf16_t* QfI,
                                              bf16_t* KfR, bf16_t* KfI,
                                              bf16_t* VfR, bf16_t* VfI) {
  __shared__ __align__(64) bf16_t Xa[EE * 32];   // [e][t] bf16 tile, 4 KB
  const int wg = blockIdx.x;
  const int tensor = wg / (BB * HH);
  const int bh = wg % (BB * HH);
  const int b = bh / HH, h = bh % HH;
  const float* src = (tensor == 0) ? q : (tensor == 1 ? k : v);
  bf16_t* dstR = (tensor == 0) ? QfR : (tensor == 1 ? KfR : VfR);
  bf16_t* dstI = (tensor == 0) ? QfI : (tensor == 1 ? KfI : VfI);

  const int tid = threadIdx.x;
  const int wave = tid >> 5, lane = tid & 31;
  const int e0 = wave * 16;                      // each wave owns a 16-row e strip

  const v8f vz = {};
  v8f aR[4], aI[4];
#pragma unroll
  for (int j = 0; j < 4; ++j) { aR[j] = vz; aI[j] = vz; }

  const int eq = (tid & 15) * 4;                 // cooperative-load coords
  const int tq = tid >> 4;                       // 0..7

  for (int kt = 0; kt < LL / 32; ++kt) {
    const int t0 = kt * 32;
    // stage [32 t][64 e] f32 -> LDS [e][t] bf16 (transpose + downconvert)
#pragma unroll
    for (int i = 0; i < 4; ++i) {
      const int t = tq + i * 8;
      const float4 f = *(const float4*)(src + (((long)b * LL + t0 + t) * HH + h) * EE + eq);
      Xa[(eq + 0) * 32 + t] = f2b(f.x);
      Xa[(eq + 1) * 32 + t] = f2b(f.y);
      Xa[(eq + 2) * 32 + t] = f2b(f.z);
      Xa[(eq + 3) * 32 + t] = f2b(f.w);
    }
    __syncthreads();
    const v16bf a = load_a_frag(Xa + e0 * 32, 32, lane);
#pragma unroll
    for (int j = 0; j < 4; ++j) {
      const v16bf bc = load_b_frag(WcT + (j * 16) * LL + t0, LL, lane);
      const v16bf bs = load_b_frag(WsT + (j * 16) * LL + t0, LL, lane);
      aR[j] = wmma_bf16(a, bc, aR[j]);
      aI[j] = wmma_bf16(a, bs, aI[j]);
    }
    __syncthreads();
  }

  const int colb = lane & 15;
  const int rowoff = (lane >> 4) * 8;
#pragma unroll
  for (int j = 0; j < 4; ++j) {
#pragma unroll
    for (int r = 0; r < 8; ++r) {
      const int e = e0 + r + rowoff;
      const int m = j * 16 + colb;
      if (tensor == 2) {                          // Vf: [e][m]
        dstR[(bh * 64 + e) * 64 + m] = f2b(aR[j][r]);
        dstI[(bh * 64 + e) * 64 + m] = f2b(aI[j][r]);
      } else {                                    // Qf/Kf: [m][e]
        dstR[(bh * 64 + m) * 64 + e] = f2b(aR[j][r]);
        dstI[(bh * 64 + m) * 64 + e] = f2b(aI[j][r]);
      }
    }
  }
}

// ------------------------------------------- QK = tanh(Qf . Kf^T over E) ----
// QK[x,y] = sum_e Qf[e,x]*Kf[e,y] (complex); Qf,Kf stored [m][e].
__global__ void __launch_bounds__(128) k_qk(const bf16_t* QfR, const bf16_t* QfI,
                                            const bf16_t* KfR, const bf16_t* KfI,
                                            bf16_t* QKr, bf16_t* QKi) {
  const int bh = blockIdx.x;
  const int tid = threadIdx.x, wave = tid >> 5, lane = tid & 31;
  const int x0 = wave * 16;
  const bf16_t* Qr = QfR + bh * 4096;
  const bf16_t* Qi = QfI + bh * 4096;
  const bf16_t* Kr = KfR + bh * 4096;
  const bf16_t* Ki = KfI + bh * 4096;

  const v8f vz = {};
  v8f rr[4], ii[4], ri[4], ir[4];
#pragma unroll
  for (int j = 0; j < 4; ++j) { rr[j] = vz; ii[j] = vz; ri[j] = vz; ir[j] = vz; }

#pragma unroll
  for (int kc = 0; kc < 64; kc += 32) {
    const v16bf aR = load_a_frag(Qr + x0 * 64 + kc, 64, lane);
    const v16bf aI = load_a_frag(Qi + x0 * 64 + kc, 64, lane);
#pragma unroll
    for (int j = 0; j < 4; ++j) {
      const v16bf bR = load_b_frag(Kr + (j * 16) * 64 + kc, 64, lane);
      const v16bf bI = load_b_frag(Ki + (j * 16) * 64 + kc, 64, lane);
      rr[j] = wmma_bf16(aR, bR, rr[j]);
      ii[j] = wmma_bf16(aI, bI, ii[j]);
      ri[j] = wmma_bf16(aR, bI, ri[j]);
      ir[j] = wmma_bf16(aI, bR, ir[j]);
    }
  }

  const int colb = lane & 15, rowoff = (lane >> 4) * 8;
#pragma unroll
  for (int j = 0; j < 4; ++j) {
#pragma unroll
    for (int r = 0; r < 8; ++r) {
      const int x = x0 + r + rowoff, y = j * 16 + colb;
      QKr[bh * 4096 + x * 64 + y] = f2b(tanhf(rr[j][r] - ii[j][r]));
      QKi[bh * 4096 + x * 64 + y] = f2b(tanhf(ri[j][r] + ir[j][r]));
    }
  }
}

// ---------------------------------------- QKV[e,x] = sum_y QK[x,y]Vf[e,y] ----
// Output stored [h][x][b][e] so stage 4's A fragment is contiguous.
__global__ void __launch_bounds__(128) k_qkv(const bf16_t* QKr, const bf16_t* QKi,
                                             const bf16_t* VfR, const bf16_t* VfI,
                                             bf16_t* Tr, bf16_t* Ti) {
  const int bh = blockIdx.x;
  const int b = bh / HH, h = bh % HH;
  const int tid = threadIdx.x, wave = tid >> 5, lane = tid & 31;
  const int x0 = wave * 16;
  const bf16_t* Ar = QKr + bh * 4096;   // [x][y]
  const bf16_t* Ai = QKi + bh * 4096;
  const bf16_t* Br = VfR + bh * 4096;   // [e][y]
  const bf16_t* Bi = VfI + bh * 4096;

  const v8f vz = {};
  v8f rr[4], ii[4], ri[4], ir[4];
#pragma unroll
  for (int j = 0; j < 4; ++j) { rr[j] = vz; ii[j] = vz; ri[j] = vz; ir[j] = vz; }

#pragma unroll
  for (int kc = 0; kc < 64; kc += 32) {
    const v16bf aR = load_a_frag(Ar + x0 * 64 + kc, 64, lane);
    const v16bf aI = load_a_frag(Ai + x0 * 64 + kc, 64, lane);
#pragma unroll
    for (int j = 0; j < 4; ++j) {
      const v16bf bR = load_b_frag(Br + (j * 16) * 64 + kc, 64, lane);
      const v16bf bI = load_b_frag(Bi + (j * 16) * 64 + kc, 64, lane);
      rr[j] = wmma_bf16(aR, bR, rr[j]);
      ii[j] = wmma_bf16(aI, bI, ii[j]);
      ri[j] = wmma_bf16(aR, bI, ri[j]);
      ir[j] = wmma_bf16(aI, bR, ir[j]);
    }
  }

  const int colb = lane & 15, rowoff = (lane >> 4) * 8;
#pragma unroll
  for (int j = 0; j < 4; ++j) {
#pragma unroll
    for (int r = 0; r < 8; ++r) {
      const int x = x0 + r + rowoff, e = j * 16 + colb;
      const int dst = ((h * 64 + x) * 16 + b) * 64 + e;
      Tr[dst] = f2b(rr[j][r] - ii[j][r]);
      Ti[dst] = f2b(ri[j][r] + ir[j][r]);
    }
  }
}

// ------------------------------- per-mode GEMM: P[b,o] = sum_e T[b,e]W[e,o] ----
// One wave per (h,x). Output stored [b][h][o][x].
__global__ void __launch_bounds__(128) k_modemm(const bf16_t* Tr, const bf16_t* Ti,
                                                const bf16_t* WtR, const bf16_t* WtI,
                                                bf16_t* Pr, bf16_t* Pi) {
  const int h = blockIdx.x >> 4;
  const int xg = blockIdx.x & 15;
  const int tid = threadIdx.x, wave = tid >> 5, lane = tid & 31;
  const int x = xg * 4 + wave;
  const bf16_t* Ar = Tr + ((h * 64 + x) * 16) * 64;   // [b][e]
  const bf16_t* Ai = Ti + ((h * 64 + x) * 16) * 64;
  const bf16_t* Br = WtR + ((h * 64 + x) * 64) * 64;  // [o][e]
  const bf16_t* Bi = WtI + ((h * 64 + x) * 64) * 64;

  const v8f vz = {};
  v8f rr[4], ii[4], ri[4], ir[4];
#pragma unroll
  for (int j = 0; j < 4; ++j) { rr[j] = vz; ii[j] = vz; ri[j] = vz; ir[j] = vz; }

#pragma unroll
  for (int kc = 0; kc < 64; kc += 32) {
    const v16bf aR = load_a_frag(Ar + kc, 64, lane);
    const v16bf aI = load_a_frag(Ai + kc, 64, lane);
#pragma unroll
    for (int j = 0; j < 4; ++j) {
      const v16bf bR = load_b_frag(Br + (j * 16) * 64 + kc, 64, lane);
      const v16bf bI = load_b_frag(Bi + (j * 16) * 64 + kc, 64, lane);
      rr[j] = wmma_bf16(aR, bR, rr[j]);
      ii[j] = wmma_bf16(aI, bI, ii[j]);
      ri[j] = wmma_bf16(aR, bI, ri[j]);
      ir[j] = wmma_bf16(aI, bR, ir[j]);
    }
  }

  const int colb = lane & 15, rowoff = (lane >> 4) * 8;
#pragma unroll
  for (int j = 0; j < 4; ++j) {
#pragma unroll
    for (int r = 0; r < 8; ++r) {
      const int bb = r + rowoff, o = j * 16 + colb;
      const int dst = ((bb * HH + h) * 64 + o) * 64 + x;
      Pr[dst] = f2b(rr[j][r] - ii[j][r]);
      Pi[dst] = f2b(ri[j][r] + ir[j][r]);
    }
  }
}

// --------------------------------------------------------- inverse DFT ----
// y[b,t,h,o] = sum_x Pre[o,x]*IC[x,t] + Pim[o,x]*IS[x,t]
__global__ void __launch_bounds__(128) k_idft(const bf16_t* Pr, const bf16_t* Pi,
                                              const bf16_t* ICt, const bf16_t* ISt,
                                              float* out) {
  const int bh = blockIdx.x;
  const int b = bh / HH, h = bh % HH;
  const int tid = threadIdx.x, wave = tid >> 5, lane = tid & 31;
  const int o0 = wave * 16;
  const bf16_t* Ar = Pr + bh * 4096;    // [o][x]
  const bf16_t* Ai = Pi + bh * 4096;

  // resident A fragments (K = 64 modes)
  const v16bf aR0 = load_a_frag(Ar + o0 * 64 + 0, 64, lane);
  const v16bf aR1 = load_a_frag(Ar + o0 * 64 + 32, 64, lane);
  const v16bf aI0 = load_a_frag(Ai + o0 * 64 + 0, 64, lane);
  const v16bf aI1 = load_a_frag(Ai + o0 * 64 + 32, 64, lane);

  const int colb = lane & 15, rowoff = (lane >> 4) * 8;
  for (int tj = 0; tj < LL / 16; ++tj) {
    const int t0 = tj * 16;
    v8f acc = {};
    acc = wmma_bf16(aR0, load_b_frag(ICt + t0 * 64 + 0, 64, lane), acc);
    acc = wmma_bf16(aR1, load_b_frag(ICt + t0 * 64 + 32, 64, lane), acc);
    acc = wmma_bf16(aI0, load_b_frag(ISt + t0 * 64 + 0, 64, lane), acc);
    acc = wmma_bf16(aI1, load_b_frag(ISt + t0 * 64 + 32, 64, lane), acc);
    const int t = t0 + colb;
#pragma unroll
    for (int r = 0; r < 8; ++r) {
      const int o = o0 + r + rowoff;
      out[(((long)b * LL + t) * HH + h) * EE + o] = acc[r];
    }
  }
}

// ------------------------------------------------------------------ host ----
extern "C" void kernel_launch(void* const* d_in, const int* in_sizes, int n_in,
                              void* d_out, int out_size, void* d_ws, size_t ws_size,
                              hipStream_t stream) {
  (void)in_sizes; (void)n_in; (void)out_size; (void)ws_size;
  const float* q  = (const float*)d_in[0];
  const float* k  = (const float*)d_in[1];
  const float* v  = (const float*)d_in[2];
  const float* wr = (const float*)d_in[3];
  const float* wi = (const float*)d_in[4];
  float* out = (float*)d_out;

  char* p = (char*)d_ws;
  auto alloc = [&](size_t bytes) -> char* {
    char* r = p; p += (bytes + 255) & ~(size_t)255; return r;
  };
  bf16_t* WcT = (bf16_t*)alloc((size_t)MM * LL * 2);            // 256 KB
  bf16_t* WsT = (bf16_t*)alloc((size_t)MM * LL * 2);
  bf16_t* ICt = (bf16_t*)alloc((size_t)LL * MM * 2);
  bf16_t* ISt = (bf16_t*)alloc((size_t)LL * MM * 2);
  bf16_t* WtR = (bf16_t*)alloc((size_t)HH * MM * EE * EE * 2);  // 4 MB
  bf16_t* WtI = (bf16_t*)alloc((size_t)HH * MM * EE * EE * 2);
  const size_t fsz = (size_t)BB * HH * EE * MM * 2;             // 1 MB each
  bf16_t* QfR = (bf16_t*)alloc(fsz); bf16_t* QfI = (bf16_t*)alloc(fsz);
  bf16_t* KfR = (bf16_t*)alloc(fsz); bf16_t* KfI = (bf16_t*)alloc(fsz);
  bf16_t* VfR = (bf16_t*)alloc(fsz); bf16_t* VfI = (bf16_t*)alloc(fsz);
  bf16_t* QKr = (bf16_t*)alloc(fsz); bf16_t* QKi = (bf16_t*)alloc(fsz);
  bf16_t* Tr  = (bf16_t*)alloc(fsz); bf16_t* Ti  = (bf16_t*)alloc(fsz);
  bf16_t* Pr  = (bf16_t*)alloc(fsz); bf16_t* Pi  = (bf16_t*)alloc(fsz);

  k_fwd_tables<<<dim3((MM * LL + 255) / 256), dim3(256), 0, stream>>>(WcT, WsT);
  k_inv_tables<<<dim3((LL * MM + 255) / 256), dim3(256), 0, stream>>>(ICt, ISt);
  k_w_transpose<<<dim3((HH * EE * EE * MM + 255) / 256), dim3(256), 0, stream>>>(wr, wi, WtR, WtI);

  k_fdft<<<dim3(BB * HH * 3), dim3(128), 0, stream>>>(q, k, v, WcT, WsT,
                                                      QfR, QfI, KfR, KfI, VfR, VfI);
  k_qk<<<dim3(BB * HH), dim3(128), 0, stream>>>(QfR, QfI, KfR, KfI, QKr, QKi);
  k_qkv<<<dim3(BB * HH), dim3(128), 0, stream>>>(QKr, QKi, VfR, VfI, Tr, Ti);
  k_modemm<<<dim3(HH * 16), dim3(128), 0, stream>>>(Tr, Ti, WtR, WtI, Pr, Pi);
  k_idft<<<dim3(BB * HH), dim3(128), 0, stream>>>(Pr, Pi, ICt, ISt, out);
}
